// foo_4733053960504
// MI455X (gfx1250) — matrix operand order repaired and checked by reading, not compiled, not run on hardware
//
#include <hip/hip_runtime.h>

typedef __attribute__((ext_vector_type(16))) __bf16 v16bf;
typedef __attribute__((ext_vector_type(8)))  __bf16 v8bf;
typedef __attribute__((ext_vector_type(4)))  __bf16 v4bf;
typedef __attribute__((ext_vector_type(8)))  float  v8f;
typedef __attribute__((ext_vector_type(4)))  unsigned int v4u;
typedef __attribute__((ext_vector_type(4)))  int v4i;
typedef __attribute__((ext_vector_type(8)))  int v8i;

#define D      64             // head dim
#define BN     128            // keys per iteration
#define NTILE  (BN / 16)      // 8 key tiles
#define NCH    (BN / 32)      // 4 A-fragment chunks for PV
#define WAVES  4              // waves per block (wave32)
#define BM     (WAVES * 16)   // query rows per block
#define NT     (WAVES * 32)   // 128 threads
#define KROW   72             // K LDS pitch (halves): 36-dword lane stride, conflict-free
#define VROW   136            // V/P LDS pitch (halves): 68-dword lane stride, conflict-free
#define PROW   136
#define DVR    (D + 16)       // V rows + ones rows (row-sum via WMMA)

#define WMMA_BF16(a, b, c) \
  __builtin_amdgcn_wmma_f32_16x16x32_bf16(false, (a), false, (b), (short)0, (c), false, false)

// ---------------- DPP16 16-lane max-reduce+broadcast ----------------
template <int CTRL>
__device__ __forceinline__ float dppmov(float x) {
  return __int_as_float(__builtin_amdgcn_update_dpp(
      __float_as_int(x), __float_as_int(x), CTRL, 0xF, 0xF, true));
}
__device__ __forceinline__ float rowmax16(float x) {
  x = fmaxf(x, dppmov<0xB1>(x));    // quad_perm [1,0,3,2]
  x = fmaxf(x, dppmov<0x4E>(x));    // quad_perm [2,3,0,1]
  x = fmaxf(x, dppmov<0x141>(x));   // row_half_mirror
  x = fmaxf(x, dppmov<0x140>(x));   // row_mirror
  return x;
}

// ---------------- 16B-aligned LDS fragment load (2 x ds_load_b128) ----------------
__device__ __forceinline__ v16bf ld_frag(const __bf16* p) {
  const v8bf a = *(const v8bf*)p;
  const v8bf b = *(const v8bf*)(p + 8);
  return __builtin_shufflevector(a, b,
      0, 1, 2, 3, 4, 5, 6, 7, 8, 9, 10, 11, 12, 13, 14, 15);
}

// ---------------- LDS byte offset of a __shared__ object ----------------
typedef __attribute__((address_space(3))) char lds_char_t;
__device__ __forceinline__ unsigned lds_off(const void* p) {
  return (unsigned)(unsigned long long)(lds_char_t*)p;
}

// ---------------- Tensor Data Mover: async 2D bf16 tile -> LDS ----------------
// pad_code: LDS pad interval code (4 = every 32 dwords, 5 = every 64 dwords),
// pad amount fixed at 4 dwords -> pitches 72 / 136 halves.
__device__ __forceinline__ void tdm_load_2d(
    unsigned lds_byte_addr, const void* gptr,
    unsigned tensor_d0, unsigned tensor_d1, unsigned stride0,
    unsigned tile_d0, unsigned tile_d1, unsigned pad_code)
{
  const unsigned long long ga = (unsigned long long)(size_t)gptr;
  v4u g0;
  g0[0] = 1u;                                            // count=1, user descriptor
  g0[1] = lds_byte_addr;                                 // lds_addr
  g0[2] = (unsigned)ga;                                  // global_addr[31:0]
  g0[3] = (unsigned)((ga >> 32) & 0x1FFFFFFu) | (2u << 30);  // addr[56:32] | type=2
  v8i g1;
  g1[0] = (int)((1u << 16)          // data_size = 2 bytes
              | (1u << 20)          // pad_enable
              | (pad_code << 22)    // pad_interval
              | (3u << 25));        // pad_amount: 4 dwords
  g1[1] = (int)((tensor_d0 & 0xFFFFu) << 16);                         // dim0[15:0]
  g1[2] = (int)((tensor_d0 >> 16) | ((tensor_d1 & 0xFFFFu) << 16));   // dim0[31:16], dim1[15:0]
  g1[3] = (int)((tensor_d1 >> 16) | (tile_d0 << 16));                 // dim1[31:16], tile_dim0
  g1[4] = (int)tile_d1;                                               // tile_dim1 (tile_dim2=0)
  g1[5] = (int)stride0;                                               // dim0_stride[31:0]
  g1[6] = 0;
  g1[7] = 0;
  v4i g2 = {0, 0, 0, 0};
  v4i g3 = {0, 0, 0, 0};
#if __clang_major__ >= 23
  v8i g4 = {0, 0, 0, 0, 0, 0, 0, 0};
  __builtin_amdgcn_tensor_load_to_lds(g0, g1, g2, g3, g4, 0);
#else
  __builtin_amdgcn_tensor_load_to_lds(g0, g1, g2, g3, 0);
#endif
}

// ---------------- pre-pass: K -> bf16 row-major, V -> bf16 transposed [d][M] ----------------
__global__ __launch_bounds__(256) void convert_kv_kernel(
    const float* __restrict__ k, const float* __restrict__ v,
    __bf16* __restrict__ kb, __bf16* __restrict__ vt, int M)
{
  const int i   = blockIdx.x * 256 + threadIdx.x;  // float4 slot
  const int idx = i << 2;
  const int row = idx >> 6;        // key
  const int col = idx & 63;        // d
  const float4 kf = *(const float4*)(k + idx);
  const float4 vf = *(const float4*)(v + idx);
  v4bf kp;
  kp[0] = (__bf16)kf.x; kp[1] = (__bf16)kf.y;
  kp[2] = (__bf16)kf.z; kp[3] = (__bf16)kf.w;
  *(v4bf*)&kb[idx] = kp;
  vt[(size_t)(col + 0) * M + row] = (__bf16)vf.x;
  vt[(size_t)(col + 1) * M + row] = (__bf16)vf.y;
  vt[(size_t)(col + 2) * M + row] = (__bf16)vf.z;
  vt[(size_t)(col + 3) * M + row] = (__bf16)vf.w;
}

// ---------------- main flash-attention kernel ----------------
__global__ __launch_bounds__(NT) void fattn_fwd_kernel(
    const float* __restrict__ q, const __bf16* __restrict__ kb,
    const __bf16* __restrict__ vt, float* __restrict__ out, int M)
{
  __shared__ __bf16 k_s[2][BN * KROW];      // [key][d]
  __shared__ __bf16 v_s[2][DVR * VROW];     // [d][key], rows D..D+15 = 1.0
  __shared__ __bf16 p_s[WAVES * 16 * PROW]; // per-wave P scratch

  const int tid  = threadIdx.x;
  const int wave = tid >> 5;
  const int lane = tid & 31;
  const int r    = lane & 15;
  const int g    = lane >> 4;

  const int qb = blockIdx.x * BM + wave * 16;
  const float qscale = 0.125f * 1.4426950408889634f;   // 1/sqrt(64) * log2(e)

  // Q tile (16x64) into WMMA A layout, pre-scaled into log2 domain
  v16bf aq0, aq1;
  {
    const float* qrow = q + (size_t)(qb + r) * D;
#pragma unroll
    for (int e = 0; e < 16; ++e) {
      const int kd = g * 16 + e;
      aq0[e] = (__bf16)(qrow[kd] * qscale);
      aq1[e] = (__bf16)(qrow[32 + kd] * qscale);
    }
  }

  // ones rows of [V | 1] (both buffers, written once; pad cols harmless)
  {
    const __bf16 one = (__bf16)1.0f;
#pragma unroll
    for (int b = 0; b < 2; ++b)
      for (int i = tid; i < 16 * VROW; i += NT)
        v_s[b][D * VROW + i] = one;
  }

  // prologue: async-stage tile 0 into buffer 0 via TDM (wave 0 only)
  if (wave == 0) {
    tdm_load_2d(lds_off(&k_s[0][0]), kb, D, (unsigned)M, D, D, BN, 4);
    tdm_load_2d(lds_off(&v_s[0][0]), vt, (unsigned)M, D, (unsigned)M, BN, D, 5);
  }

  v8f acc0 = {}, acc1 = {}, acc2 = {}, acc3 = {}, acc4 = {};
  float mrow[8];
#pragma unroll
  for (int j = 0; j < 8; ++j) mrow[j] = -1e30f;

  const int niter = M / BN;
  for (int it = 0; it < niter; ++it) {
    __builtin_amdgcn_s_wait_tensorcnt(0);   // wave0: tile `it` landed; others: no-op
    __syncthreads();                        // publish LDS to all waves
    const int cur = it & 1;

    // issue next tile's DMA; overlaps with this tile's WMMA work
    if (wave == 0 && it + 1 < niter) {
      const int nxt = cur ^ 1;
      tdm_load_2d(lds_off(&k_s[nxt][0]), kb + (size_t)(it + 1) * BN * D,
                  D, (unsigned)M, D, D, BN, 4);
      tdm_load_2d(lds_off(&v_s[nxt][0]), vt + (size_t)(it + 1) * BN,
                  (unsigned)M, D, (unsigned)M, BN, D, 5);
    }

    // ---- scores: S = Q K^T, 8 key-tiles x 2 d-chunks = 16 WMMAs ----
    const __bf16* ks = &k_s[cur][r * KROW + g * 16];
    v8f s[NTILE];
#pragma unroll
    for (int t = 0; t < NTILE; ++t) {
      const __bf16* kt = ks + t * 16 * KROW;
      v8f z = {};
      z = WMMA_BF16(aq0, ld_frag(kt), z);
      z = WMMA_BF16(aq1, ld_frag(kt + 32), z);
      s[t] = z;
    }

    // ---- online softmax in base-2 domain ----
    __bf16* pw = &p_s[wave * 16 * PROW];
#pragma unroll
    for (int j = 0; j < 8; ++j) {
      float x[NTILE];
#pragma unroll
      for (int t = 0; t < NTILE; ++t) x[t] = s[t][j];
      float mx = fmaxf(fmaxf(fmaxf(x[0], x[1]), fmaxf(x[2], x[3])),
                       fmaxf(fmaxf(x[4], x[5]), fmaxf(x[6], x[7])));
      mx = rowmax16(mx);
      const float mnew  = fmaxf(mrow[j], mx);
      const float alpha = exp2f(mrow[j] - mnew);
      mrow[j] = mnew;
      acc0[j] *= alpha; acc1[j] *= alpha; acc2[j] *= alpha;
      acc3[j] *= alpha; acc4[j] *= alpha;
      const int prow = (j + 8 * g) * PROW;
#pragma unroll
      for (int t = 0; t < NTILE; ++t)
        pw[prow + t * 16 + r] = (__bf16)exp2f(x[t] - mnew);
    }

    // ---- reload P in A layout (four 16x32 chunks) ----
    const __bf16* pr = &pw[r * PROW];
    v16bf ap[NCH];
#pragma unroll
    for (int c = 0; c < NCH; ++c)
      ap[c] = __builtin_shufflevector(
          *(const v8bf*)&pr[c * 32 + g * 16], *(const v8bf*)&pr[c * 32 + g * 16 + 8],
          0, 1, 2, 3, 4, 5, 6, 7, 8, 9, 10, 11, 12, 13, 14, 15);

    // ---- O += P @ [V | 1] : 5 chunks x 4 K-chunks = 20 WMMAs ----
    const __bf16* vs = &v_s[cur][r * VROW + g * 16];
#pragma unroll
    for (int c = 0; c < NCH; ++c) {
      const int co = c * 32;
      acc0 = WMMA_BF16(ap[c], ld_frag(vs + co            ), acc0);
      acc1 = WMMA_BF16(ap[c], ld_frag(vs + co + 16 * VROW), acc1);
      acc2 = WMMA_BF16(ap[c], ld_frag(vs + co + 32 * VROW), acc2);
      acc3 = WMMA_BF16(ap[c], ld_frag(vs + co + 48 * VROW), acc3);
      acc4 = WMMA_BF16(ap[c], ld_frag(vs + co + 64 * VROW), acc4);   // running row-sum
    }
  }

  // ---- epilogue: divide by row sum (replicated across lanes in acc4) ----
#pragma unroll
  for (int j = 0; j < 8; ++j) {
    const float inv = 1.0f / acc4[j];
    float* o = out + (size_t)(qb + j + 8 * g) * D + r;
    o[ 0] = acc0[j] * inv;
    o[16] = acc1[j] * inv;
    o[32] = acc2[j] * inv;
    o[48] = acc3[j] * inv;
  }
}

extern "C" void kernel_launch(void* const* d_in, const int* in_sizes, int n_in,
                              void* d_out, int out_size, void* d_ws, size_t ws_size,
                              hipStream_t stream) {
  (void)n_in; (void)out_size; (void)ws_size;   // requires ws_size >= 2*M*D*2 = 2 MB
  const float* q = (const float*)d_in[0];
  const float* k = (const float*)d_in[1];
  const float* v = (const float*)d_in[2];
  float* out = (float*)d_out;
  const int N = in_sizes[0] / D;   // 8192 query rows
  const int M = in_sizes[1] / D;   // 8192 key rows

  __bf16* kb = (__bf16*)d_ws;           // K as bf16, row-major [M][D]
  __bf16* vt = kb + (size_t)M * D;      // V as bf16, transposed [D][M]

  convert_kv_kernel<<<dim3((M * D / 4) / 256), dim3(256), 0, stream>>>(k, v, kb, vt, M);
  fattn_fwd_kernel<<<dim3(N / BM), dim3(NT), 0, stream>>>(q, kb, vt, out, M);
}
